// UpperGraphCritic_90683939487814
// MI455X (gfx1250) — matrix operand order repaired and checked by reading, not compiled, not run on hardware
//
#include <hip/hip_runtime.h>
#include <hip/hip_bf16.h>

#define N_AGENTS 8
#define OBS 64
#define HID 128
#define ATT 64
#define FIN (HID + ATT)   // 192
#define ROWS_PER_BLK 64
#define THREADS 128

typedef __attribute__((ext_vector_type(16))) _Float16 v16h;
typedef __attribute__((ext_vector_type(8)))  _Float16 v8h;
typedef __attribute__((ext_vector_type(4)))  _Float16 v4h;
typedef __attribute__((ext_vector_type(8)))  float    v8f;
typedef __attribute__((ext_vector_type(4)))  float    v4f;
typedef __attribute__((ext_vector_type(4)))  int      v4i;

// ---- weight workspace layout (f16 elements) ----
#define W_ENC 0          // 128x64
#define W_IH  8192       // 384x128
#define W_HH  57344      // 384x128
#define W_Q   106496     // 64x128
#define W_K   114688     // 64x128
#define W_V   122880     // 64x128
#define W_RL1 131072     // 128x192
#define W_RG1 155648     // 128x192
#define W_TOTAL 180224

// A-matrix fragment (16x32 f16), per ISA 7.12.2
__device__ inline v16h frag_a_h(const _Float16* base, int stride, int m, int hi, int k0) {
  const _Float16* p = base + m * stride + k0 + hi * 8;
  v8h lo = *(const v8h*)p;
  v8h hh = *(const v8h*)(p + 16);
  v16h r;
#pragma unroll
  for (int i = 0; i < 8; ++i) { r[i] = lo[i]; r[i + 8] = hh[i]; }
  return r;
}

__device__ inline v16h frag_a_f32(const float* base, int stride, int m, int hi, int k0) {
  const float* p = base + (size_t)m * stride + k0 + hi * 8;
  v4f a0 = *(const v4f*)p;
  v4f a1 = *(const v4f*)(p + 4);
  v4f a2 = *(const v4f*)(p + 16);
  v4f a3 = *(const v4f*)(p + 20);
  v16h r;
#pragma unroll
  for (int i = 0; i < 4; ++i) {
    r[i]      = (_Float16)a0[i];
    r[i + 4]  = (_Float16)a1[i];
    r[i + 8]  = (_Float16)a2[i];
    r[i + 12] = (_Float16)a3[i];
  }
  return r;
}

// B-matrix fragment (32x16 f16) read from row-major [N][K] weights
__device__ inline v16h frag_b_h(const _Float16* w, int strideK, int n, int hi, int n0, int k0) {
  const _Float16* p = w + (size_t)(n0 + n) * strideK + k0 + hi * 8;
  v8h lo = *(const v8h*)p;
  v8h hh = *(const v8h*)(p + 16);
  v16h r;
#pragma unroll
  for (int i = 0; i < 8; ++i) { r[i] = lo[i]; r[i + 8] = hh[i]; }
  return r;
}

__device__ inline v8f wmma_f16(v16h a, v16h b, v8f c) {
  return __builtin_amdgcn_wmma_f32_16x16x32_f16(false, a, false, b, (short)0, c, false, false);
}

__device__ inline float sigmoidf_(float x) { return 1.0f / (1.0f + __expf(-x)); }

// ---------------- weight f32 -> f16 conversion ----------------
__global__ void cvt_weights_kernel(const float* enc, const float* wih, const float* whh,
                                   const float* qw, const float* kw, const float* vw,
                                   const float* rl1, const float* rg1, _Float16* W) {
  int i = blockIdx.x * blockDim.x + threadIdx.x;
  if (i >= W_TOTAL) return;
  int o = i;
  float v;
  if (o < 8192) v = enc[o];
  else if ((o -= 8192) < 49152) v = wih[o];
  else if ((o -= 49152) < 49152) v = whh[o];
  else if ((o -= 49152) < 8192) v = qw[o];
  else if ((o -= 8192) < 8192) v = kw[o];
  else if ((o -= 8192) < 8192) v = vw[o];
  else if ((o -= 8192) < 24576) v = rl1[o];
  else { o -= 24576; v = rg1[o]; }
  W[i] = (_Float16)v;
}

// ---------------- fused critic kernel ----------------
__launch_bounds__(THREADS, 2)
__global__ void critic_kernel(const float* __restrict__ X, const float* __restrict__ Hin,
                              const _Float16* __restrict__ W,
                              const float* __restrict__ enc_b,
                              const float* __restrict__ b_ih, const float* __restrict__ b_hh,
                              const float* __restrict__ v_b,
                              const float* __restrict__ rl1_b,
                              const float* __restrict__ rl2_w, const float* __restrict__ rl2_b,
                              const float* __restrict__ rg1_b,
                              const float* __restrict__ rg2_w, const float* __restrict__ rg2_b,
                              float* __restrict__ outVg, float* __restrict__ outVl,
                              float* __restrict__ outH) {
  // region A (32 KB) is phase-overlaid; fin (24 KB) lives whole kernel.
  __shared__ __align__(16) _Float16 smemA[16384];
  __shared__ __align__(16) _Float16 fin[ROWS_PER_BLK * FIN];

  const int tid  = threadIdx.x;
  const int lane = tid & 31;
  const int wave = tid >> 5;
  const int m    = lane & 15;
  const int hi   = lane >> 4;
  const int rowBase = blockIdx.x * ROWS_PER_BLK;
  const int r0  = rowBase + wave * 16;   // global row of this wave's tile
  const int lr0 = wave * 16;             // local row

  // warm the L2-resident f16 weights (global_prefetch_b8 path)
  __builtin_prefetch((const char*)W + tid * 128, 0, 1);

  const _Float16* Wenc = W + W_ENC;
  const _Float16* Wih  = W + W_IH;
  const _Float16* Whh  = W + W_HH;
  const _Float16* Wq   = W + W_Q;
  const _Float16* Wk   = W + W_K;
  const _Float16* Wv   = W + W_V;
  const _Float16* Wrl1 = W + W_RL1;
  const _Float16* Wrg1 = W + W_RG1;

  // phase 0 overlay: Hin f32 staging tile (32 KB, whole of region A)
  float* hstage = (float*)smemA;          // [64][128] f32
  // phase 1 overlay
  _Float16* eAct = smemA;                 // [64][128]
  // phase 2 overlay
  _Float16* q16  = smemA;                 // [64][64]
  _Float16* k16  = smemA + 4096;          // [64][64]
  _Float16* v16l = smemA + 8192;          // [64][64]
  // phase 3 overlay
  _Float16* hl16 = smemA;                 // [64][128]
  _Float16* gf16 = smemA + 8192;          // [16][192] (rows 8..15 zero pad)
  _Float16* hg16 = smemA + 11264;         // [16][128]

  // ---- stage 0: stage Hin tile to LDS once; keep f16 copy in fin cols 0..127 ----
  {
    const float* gsrc = Hin + (size_t)rowBase * HID;
#if defined(__has_builtin) && __has_builtin(__builtin_amdgcn_global_load_async_to_lds_b128)
    // ASYNCcnt-tracked copy: global -> LDS, 16 B per lane per op (ISA 08 4.2)
    typedef __attribute__((address_space(1))) v4i gv4i;
    typedef __attribute__((address_space(3))) v4i lv4i;
#pragma unroll
    for (int i = 0; i < 16; ++i) {
      int idx4 = (tid + i * THREADS) * 4;
      __builtin_amdgcn_global_load_async_to_lds_b128(
          (gv4i*)(gsrc + idx4), (lv4i*)(hstage + idx4), 0, 0);
    }
#if __has_builtin(__builtin_amdgcn_s_wait_asynccnt)
    __builtin_amdgcn_s_wait_asynccnt(0);
#else
    asm volatile("s_wait_asynccnt 0x0" ::: "memory");
#endif
    __syncthreads();
#pragma unroll
    for (int i = 0; i < 16; ++i) {
      int idx4 = (tid + i * THREADS) * 4;
      int row = idx4 / HID, col = idx4 % HID;
      v4f hv = *(const v4f*)(hstage + idx4);
      v4h hh;
#pragma unroll
      for (int j = 0; j < 4; ++j) hh[j] = (_Float16)hv[j];
      *(v4h*)(fin + row * FIN + col) = hh;
    }
#else
    // fallback: convert straight from global
#pragma unroll
    for (int i = 0; i < 16; ++i) {
      int idx4 = (tid + i * THREADS) * 4;
      int row = idx4 / HID, col = idx4 % HID;
      v4f hv = *(const v4f*)(gsrc + idx4);
      v4h hh;
#pragma unroll
      for (int j = 0; j < 4; ++j) hh[j] = (_Float16)hv[j];
      *(v4h*)(fin + row * FIN + col) = hh;
    }
#endif
  }
  __syncthreads();   // hstage dead; region A becomes eAct

  // ---- stage 1: e = relu(X @ enc_w^T + enc_b) ----
  {
    v16h aX[2];
#pragma unroll
    for (int ks = 0; ks < 2; ++ks)
      aX[ks] = frag_a_f32(X + (size_t)r0 * OBS, OBS, m, hi, ks * 32);
    for (int c0 = 0; c0 < 8; ++c0) {
      v8f acc = {};
#pragma unroll
      for (int ks = 0; ks < 2; ++ks)
        acc = wmma_f16(aX[ks], frag_b_h(Wenc, OBS, m, hi, c0 * 16, ks * 32), acc);
      const int col = c0 * 16 + m;
      const float bb = enc_b[col];
#pragma unroll
      for (int v = 0; v < 8; ++v) {
        float x = acc[v] + bb;
        x = x > 0.f ? x : 0.f;
        eAct[(lr0 + v + 8 * hi) * HID + col] = (_Float16)x;
      }
    }
  }
  // wave-private rows: no block barrier needed before GRU

  // ---- stage 2: GRU cell -> h_out (global f32 + fin f16) ----
  {
    v16h aE[4], aH[4];
#pragma unroll
    for (int ks = 0; ks < 4; ++ks) {
      aE[ks] = frag_a_h(eAct + lr0 * HID, HID, m, hi, ks * 32);
      aH[ks] = frag_a_h(fin + lr0 * FIN, FIN, m, hi, ks * 32);   // h_prev (f16)
    }
    for (int c0 = 0; c0 < 8; ++c0) {
      v8f aIR = {}, aIZ = {}, aIN = {}, aHR = {}, aHZ = {}, aHN = {};
#pragma unroll
      for (int ks = 0; ks < 4; ++ks) {
        const int k0 = ks * 32;
        aIR = wmma_f16(aE[ks], frag_b_h(Wih,             HID, m, hi, c0 * 16, k0), aIR);
        aIZ = wmma_f16(aE[ks], frag_b_h(Wih + 128 * HID, HID, m, hi, c0 * 16, k0), aIZ);
        aIN = wmma_f16(aE[ks], frag_b_h(Wih + 256 * HID, HID, m, hi, c0 * 16, k0), aIN);
        aHR = wmma_f16(aH[ks], frag_b_h(Whh,             HID, m, hi, c0 * 16, k0), aHR);
        aHZ = wmma_f16(aH[ks], frag_b_h(Whh + 128 * HID, HID, m, hi, c0 * 16, k0), aHZ);
        aHN = wmma_f16(aH[ks], frag_b_h(Whh + 256 * HID, HID, m, hi, c0 * 16, k0), aHN);
      }
      const int col = c0 * 16 + m;
      const float bir = b_ih[col], biz = b_ih[128 + col], bin = b_ih[256 + col];
      const float bhr = b_hh[col], bhz = b_hh[128 + col], bhn = b_hh[256 + col];
#pragma unroll
      for (int v = 0; v < 8; ++v) {
        const int row = v + 8 * hi;
        float r  = sigmoidf_(aIR[v] + bir + aHR[v] + bhr);
        float z  = sigmoidf_(aIZ[v] + biz + aHZ[v] + bhz);
        float nn = tanhf(aIN[v] + bin + r * (aHN[v] + bhn));
        float hp = (float)fin[(lr0 + row) * FIN + col];  // h_prev from LDS (read-before-write)
        float ho = (1.f - z) * nn + z * hp;
        outH[(size_t)(r0 + row) * HID + col] = ho;
        fin[(lr0 + row) * FIN + col] = (_Float16)ho;
      }
    }
  }
  __syncthreads();   // eAct dead; region A becomes q/k/v

  // ---- stage 3: q, k = h @ W^T (no bias/relu); v = relu(h @ v_w^T + v_b) ----
  {
    v16h aF[4];
#pragma unroll
    for (int ks = 0; ks < 4; ++ks)
      aF[ks] = frag_a_h(fin + lr0 * FIN, FIN, m, hi, ks * 32);
    for (int c0 = 0; c0 < 4; ++c0) {
      v8f aq = {}, ak = {}, av = {};
#pragma unroll
      for (int ks = 0; ks < 4; ++ks) {
        const int k0 = ks * 32;
        aq = wmma_f16(aF[ks], frag_b_h(Wq, HID, m, hi, c0 * 16, k0), aq);
        ak = wmma_f16(aF[ks], frag_b_h(Wk, HID, m, hi, c0 * 16, k0), ak);
        av = wmma_f16(aF[ks], frag_b_h(Wv, HID, m, hi, c0 * 16, k0), av);
      }
      const int col = c0 * 16 + m;
      const float vb = v_b[col];
#pragma unroll
      for (int v = 0; v < 8; ++v) {
        const int row = lr0 + v + 8 * hi;
        q16[row * ATT + col] = (_Float16)aq[v];
        k16[row * ATT + col] = (_Float16)ak[v];
        float xv = av[v] + vb;
        xv = xv > 0.f ? xv : 0.f;
        v16l[row * ATT + col] = (_Float16)xv;
      }
    }
  }
  __syncthreads();

  // ---- stage 4: exclude-self attention over 8-agent groups (block-local) ----
  if (tid < ROWS_PER_BLK) {
    const int lr = tid, gb = (lr / 8) * 8, i = lr % 8;
    float sc[8], mx = -1e30f;
    for (int j = 0; j < 8; ++j) {
      if (j == i) { sc[j] = -1e30f; continue; }
      float s = 0.f;
      for (int a = 0; a < ATT; ++a)
        s += (float)q16[lr * ATT + a] * (float)k16[(gb + j) * ATT + a];
      sc[j] = s * 0.125f;  // 1/sqrt(64)
      mx = fmaxf(mx, sc[j]);
    }
    float w[8], se = 0.f;
    for (int j = 0; j < 8; ++j) {
      w[j] = (j == i) ? 0.f : __expf(sc[j] - mx);
      se += w[j];
    }
    const float inv = 1.f / se;
    for (int a = 0; a < ATT; ++a) {
      float x = 0.f;
      for (int j = 0; j < 8; ++j) x += w[j] * (float)v16l[(gb + j) * ATT + a];
      fin[lr * FIN + HID + a] = (_Float16)(x * inv);
    }
  }
  __syncthreads();   // q/k/v dead; region A becomes hl/gf/hg

  // ---- stage 5: local head hidden hl = relu(fin @ rl1_w^T + b) ----
  {
    v16h aF[6];
#pragma unroll
    for (int ks = 0; ks < 6; ++ks)
      aF[ks] = frag_a_h(fin + lr0 * FIN, FIN, m, hi, ks * 32);
    for (int c0 = 0; c0 < 8; ++c0) {
      v8f acc = {};
#pragma unroll
      for (int ks = 0; ks < 6; ++ks)
        acc = wmma_f16(aF[ks], frag_b_h(Wrl1, FIN, m, hi, c0 * 16, ks * 32), acc);
      const int col = c0 * 16 + m;
      const float bb = rl1_b[col];
#pragma unroll
      for (int v = 0; v < 8; ++v) {
        float x = acc[v] + bb;
        x = x > 0.f ? x : 0.f;
        hl16[(lr0 + v + 8 * hi) * HID + col] = (_Float16)x;
      }
    }
  }
  // global feature: mean over 8 agents of fin, 8 groups x 192 (pad rows 8..15 with 0)
  for (int idx = tid; idx < 8 * FIN; idx += THREADS) {
    const int g = idx / FIN, c = idx % FIN;
    float s = 0.f;
    for (int j = 0; j < 8; ++j) s += (float)fin[(g * 8 + j) * FIN + c];
    gf16[g * FIN + c] = (_Float16)(s * 0.125f);
  }
  for (int idx = tid; idx < 8 * FIN; idx += THREADS)
    gf16[8 * FIN + idx] = (_Float16)0.f;
  __syncthreads();

  // ---- stage 6a: V_local ----
  if (tid < ROWS_PER_BLK) {
    float s = rl2_b[0];
    for (int c = 0; c < HID; ++c) s += (float)hl16[tid * HID + c] * rl2_w[c];
    outVl[rowBase + tid] = s;
  }
  // ---- stage 6b: global head (single 16-row tile; wave 0, full EXEC) ----
  if (wave == 0) {
    v16h aG[6];
#pragma unroll
    for (int ks = 0; ks < 6; ++ks)
      aG[ks] = frag_a_h(gf16, FIN, m, hi, ks * 32);
    for (int c0 = 0; c0 < 8; ++c0) {
      v8f acc = {};
#pragma unroll
      for (int ks = 0; ks < 6; ++ks)
        acc = wmma_f16(aG[ks], frag_b_h(Wrg1, FIN, m, hi, c0 * 16, ks * 32), acc);
      const int col = c0 * 16 + m;
      const float bb = rg1_b[col];
#pragma unroll
      for (int v = 0; v < 8; ++v) {
        float x = acc[v] + bb;
        x = x > 0.f ? x : 0.f;
        hg16[(v + 8 * hi) * HID + col] = (_Float16)x;
      }
    }
  }
  __syncthreads();
  if (tid < 8) {
    float s = rg2_b[0];
    for (int c = 0; c < HID; ++c) s += (float)hg16[tid * HID + c] * rg2_w[c];
    outVg[blockIdx.x * 8 + tid] = s;
  }
}

extern "C" void kernel_launch(void* const* d_in, const int* in_sizes, int n_in,
                              void* d_out, int out_size, void* d_ws, size_t ws_size,
                              hipStream_t stream) {
  const float* X     = (const float*)d_in[0];
  const float* Hin   = (const float*)d_in[2];
  const float* enc_w = (const float*)d_in[3];
  const float* enc_b = (const float*)d_in[4];
  const float* w_ih  = (const float*)d_in[5];
  const float* w_hh  = (const float*)d_in[6];
  const float* b_ih  = (const float*)d_in[7];
  const float* b_hh  = (const float*)d_in[8];
  const float* q_w   = (const float*)d_in[9];
  const float* k_w   = (const float*)d_in[10];
  const float* v_w   = (const float*)d_in[11];
  const float* v_b   = (const float*)d_in[12];
  const float* rg1_w = (const float*)d_in[13];
  const float* rg1_b = (const float*)d_in[14];
  const float* rg2_w = (const float*)d_in[15];
  const float* rg2_b = (const float*)d_in[16];
  const float* rl1_w = (const float*)d_in[17];
  const float* rl1_b = (const float*)d_in[18];
  const float* rl2_w = (const float*)d_in[19];
  const float* rl2_b = (const float*)d_in[20];

  _Float16* W = (_Float16*)d_ws;

  const int SIZE = 65536 * N_AGENTS;   // 524288 rows
  float* out   = (float*)d_out;
  float* outVg = out;                       // [65536]
  float* outVl = out + 65536;               // [524288]
  float* outH  = out + 65536 + SIZE;        // [524288*128]

  cvt_weights_kernel<<<(W_TOTAL + 255) / 256, 256, 0, stream>>>(
      enc_w, w_ih, w_hh, q_w, k_w, v_w, rl1_w, rg1_w, W);

  critic_kernel<<<SIZE / ROWS_PER_BLK, THREADS, 0, stream>>>(
      X, Hin, W, enc_b, b_ih, b_hh, v_b, rl1_b, rl2_w, rl2_b,
      rg1_b, rg2_w, rg2_b, outVg, outVl, outH);
}
